// Head_57750130262130
// MI455X (gfx1250) — matrix operand order repaired and checked by reading, not compiled, not run on hardware
//
#include <hip/hip_runtime.h>
#include <hip/hip_bf16.h>

typedef __attribute__((ext_vector_type(16))) _Float16 v16h;
typedef __attribute__((ext_vector_type(8)))  _Float16 v8h;
typedef __attribute__((ext_vector_type(8)))  float    v8f;
typedef __attribute__((ext_vector_type(4)))  float    v4f;

#define EMBED 1024
#define HEAD  64
#define CTX   2048
#define BATCH 16

static __device__ __forceinline__ v8f wmma_f16(v16h a, v16h b, v8f c) {
    // D = A(16x32 f16) * B(32x16 f16) + C(16x16 f32)
    return __builtin_amdgcn_wmma_f32_16x16x32_f16(
        /*neg_a=*/false, a, /*neg_b=*/false, b,
        /*c_mod=*/(short)0, c, /*reuse_a=*/false, /*reuse_b=*/false);
}

static __device__ __forceinline__ v8f zero8f() {
    v8f z;
#pragma unroll
    for (int i = 0; i < 8; ++i) z[i] = 0.0f;
    return z;
}

// A-matrix 16x32 f16: lane holds row m = lane&15; K contiguous run of 16:
// K = 2*kb8..2*kb8+15 (elems 0..15), kb8 = 8*(lane>=16).
static __device__ __forceinline__ v16h ld_a16(const _Float16* p, int kb8) {
    v8h lo = *(const v8h*)(p + 2 * kb8);
    v8h hi = *(const v8h*)(p + 2 * kb8 + 8);
    v16h r;
#pragma unroll
    for (int i = 0; i < 8; ++i) { r[i] = lo[i]; r[8 + i] = hi[i]; }
    return r;
}

// B-matrix 32x16 f16: lane holds column n = lane&15; halves = K contiguous run
// of 16 starting at 16*(lane>=16).  p points at that run.
static __device__ __forceinline__ v16h ld_b16(const _Float16* p) {
    v8h lo = *(const v8h*)p;
    v8h hi = *(const v8h*)(p + 8);
    v16h r;
#pragma unroll
    for (int i = 0; i < 8; ++i) { r[i] = lo[i]; r[8 + i] = hi[i]; }
    return r;
}

// ---------------------------------------------------------------------------
// Kernel 0: convert/transpose weights: Wt[w][h][e] (f16) = W_w[e][h] (f32)
// ---------------------------------------------------------------------------
__global__ void prep_weights(const float* __restrict__ Wq,
                             const float* __restrict__ Wk,
                             const float* __restrict__ Wv,
                             _Float16* __restrict__ Wt) {
    int t = blockIdx.x * blockDim.x + threadIdx.x;   // 0 .. 3*64*1024-1
    int w = t >> 16;
    int i = t & 65535;
    int h = i >> 10;          // 0..63
    int e = i & 1023;         // 0..1023
    const float* W = (w == 0) ? Wq : ((w == 1) ? Wk : Wv);
    Wt[(size_t)w * 65536 + (size_t)h * 1024 + e] = (_Float16)W[(size_t)e * 64 + h];
}

// ---------------------------------------------------------------------------
// Kernel 1: projection.  One wave = one weight matrix x one 64-row tile.
// Per K-step: 4 B tiles (kept in regs) x 4 M sub-tiles = 16 WMMAs, 24 loads.
// A-tile conversions are double-buffered (areg[ms&1]) so the f16 converts for
// sub-tile ms+1 co-execute with the WMMAs of ms (no WMMA->VALU WAR nops).
//   Qh, Kh: f16 row-major [B][S][64];  Vt: f16 transposed [B][64][S]
// ---------------------------------------------------------------------------
__global__ __launch_bounds__(32) void proj_kernel(const float* __restrict__ x,
                                                  const _Float16* __restrict__ Wt,
                                                  _Float16* __restrict__ Qh,
                                                  _Float16* __restrict__ Kh,
                                                  _Float16* __restrict__ Vt) {
    const int lane    = threadIdx.x & 31;
    const int half_id = lane >> 4;
    const int ln      = lane & 15;
    const int kb8     = half_id * 8;

    int t = blockIdx.x;               // 0..1535
    const int w    = t >> 9;          // weight matrix 0..2
    t &= 511;
    const int b    = t >> 5;          // batch 0..15
    const int mt   = t & 31;          // 64-row tile 0..31
    const int row0 = mt * 64;

    const _Float16* Wb  = Wt + (size_t)w * 65536;
    const float*    xb  = x + ((size_t)b * CTX + row0 + ln) * EMBED;  // A row m=ln

    v8f acc[4][4];                    // [m sub-tile][n tile]
#pragma unroll
    for (int ms = 0; ms < 4; ++ms)
#pragma unroll
        for (int nt = 0; nt < 4; ++nt) acc[ms][nt] = zero8f();

    v16h areg[2];                     // double-buffered A tiles

#pragma unroll 1
    for (int ks = 0; ks < 32; ++ks) {
        // ---- 4 B tiles for this K-step, held in registers ----
        v16h bb[4];
#pragma unroll
        for (int nt = 0; nt < 4; ++nt)
            bb[nt] = ld_b16(Wb + (size_t)(nt * 16 + ln) * 1024 + ks * 32 + half_id * 16);

        // ---- 4 A sub-tiles, each feeds 4 WMMAs ----
#pragma unroll
        for (int ms = 0; ms < 4; ++ms) {
            const float* ap = xb + (size_t)(ms * 16) * EMBED + ks * 32 + half_id * 16;
            v4f f0 = *(const v4f*)(ap);
            v4f f1 = *(const v4f*)(ap + 4);
            v4f f2 = *(const v4f*)(ap + 8);
            v4f f3 = *(const v4f*)(ap + 12);
            v16h& a = areg[ms & 1];
#pragma unroll
            for (int i = 0; i < 4; ++i) {
                a[i]      = (_Float16)f0[i];
                a[4 + i]  = (_Float16)f1[i];
                a[8 + i]  = (_Float16)f2[i];
                a[12 + i] = (_Float16)f3[i];
            }
#pragma unroll
            for (int nt = 0; nt < 4; ++nt)
                acc[ms][nt] = wmma_f16(a, bb[nt], acc[ms][nt]);
        }
    }

    // ---- stores ----
    if (w == 2) {
        // V transposed: packed b128 stores (8 consecutive rows at fixed h)
#pragma unroll
        for (int ms = 0; ms < 4; ++ms)
#pragma unroll
            for (int nt = 0; nt < 4; ++nt) {
                int h = nt * 16 + ln;
                v8h pv;
#pragma unroll
                for (int j = 0; j < 8; ++j) pv[j] = (_Float16)acc[ms][nt][j];
                *(v8h*)(Vt + ((size_t)b * HEAD + h) * CTX + row0 + ms * 16 + half_id * 8) = pv;
            }
    } else {
        _Float16* Oh = (w == 0) ? Qh : Kh;
        const size_t obase = ((size_t)b * CTX + row0) * HEAD;
#pragma unroll
        for (int ms = 0; ms < 4; ++ms)
#pragma unroll
            for (int nt = 0; nt < 4; ++nt) {
                int h = nt * 16 + ln;
#pragma unroll
                for (int j = 0; j < 8; ++j) {
                    int r = ms * 16 + j + half_id * 8;
                    Oh[obase + (size_t)r * HEAD + h] = (_Float16)acc[ms][nt][j];
                }
            }
    }
}

// ---------------------------------------------------------------------------
// Kernel 2: causal flash attention.  One wave = one 32-query tile (two 16-row
// sub-tiles u=0,1 sharing every K and V tile -> 16 loads : 16 WMMAs per block).
// ---------------------------------------------------------------------------
__global__ __launch_bounds__(32) void attn_kernel(const _Float16* __restrict__ Qh,
                                                  const _Float16* __restrict__ Kh,
                                                  const _Float16* __restrict__ Vt,
                                                  float* __restrict__ out) {
    __shared__ __align__(16) _Float16 pshare[16 * 32];

    const int lane    = threadIdx.x & 31;
    const int half_id = lane >> 4;
    const int ln      = lane & 15;
    const int kb8     = half_id * 8;

    const int tile = blockIdx.x;      // 0..1023
    const int b    = tile >> 6;
    const int qt   = tile & 63;
    const int q0   = qt * 32;

    // Q: two sub-tiles x two head-dim halves as A tiles
    v16h aq[2][2];
#pragma unroll
    for (int u = 0; u < 2; ++u) {
        const _Float16* qp = Qh + ((size_t)b * CTX + q0 + u * 16 + ln) * HEAD;
        aq[u][0] = ld_a16(qp, kb8);
        aq[u][1] = ld_a16(qp + 32, kb8);
    }

    float mrow[2][8], lrow[2][8];
    v8f o[2][4];
#pragma unroll
    for (int u = 0; u < 2; ++u) {
#pragma unroll
        for (int j = 0; j < 8; ++j) { mrow[u][j] = -1e30f; lrow[u][j] = 0.0f; }
#pragma unroll
        for (int nt = 0; nt < 4; ++nt) o[u][nt] = zero8f();
    }

    const int nkb = (q0 >> 5) + 1;    // key blocks of 32 (causal)
    v16h pa[2];

#pragma unroll 1
    for (int kb = 0; kb < nkb; ++kb) {
        const int key0 = kb * 32;

        // ---- scores: each K tile pair feeds both query sub-tiles ----
        v8f s[2][2];
#pragma unroll
        for (int g = 0; g < 2; ++g) {
            const _Float16* kp = Kh + ((size_t)b * CTX + key0 + g * 16 + ln) * HEAD + half_id * 16;
            v16h bk0 = ld_b16(kp);        // head dims  0..31 slice
            v16h bk1 = ld_b16(kp + 32);   // head dims 32..63 slice
#pragma unroll
            for (int u = 0; u < 2; ++u) {
                v8f z = zero8f();
                z = wmma_f16(aq[u][0], bk0, z);
                z = wmma_f16(aq[u][1], bk1, z);
                s[u][g] = z;
            }
        }

        // ---- per sub-tile: mask, online softmax, P -> A-layout via LDS ----
#pragma unroll
        for (int u = 0; u < 2; ++u) {
            float p0[8], p1[8];
#pragma unroll
            for (int j = 0; j < 8; ++j) {
                const int qr = q0 + u * 16 + j + half_id * 8;
                float s0 = s[u][0][j] * 0.125f;             // * HEAD^-0.5
                float s1 = s[u][1][j] * 0.125f;
                s0 = (key0 + ln      <= qr) ? s0 : -1e30f;
                s1 = (key0 + 16 + ln <= qr) ? s1 : -1e30f;

                float mx = fmaxf(s0, s1);
#pragma unroll
                for (int d = 1; d < 16; d <<= 1) mx = fmaxf(mx, __shfl_xor(mx, d, 32));
                const float mnew = fmaxf(mrow[u][j], mx);
                const float corr = __expf(mrow[u][j] - mnew);
                const float e0 = __expf(s0 - mnew);
                const float e1 = __expf(s1 - mnew);
                float rs = e0 + e1;
#pragma unroll
                for (int d = 1; d < 16; d <<= 1) rs += __shfl_xor(rs, d, 32);

                lrow[u][j] = lrow[u][j] * corr + rs;
                mrow[u][j] = mnew;
                p0[j] = e0; p1[j] = e1;
#pragma unroll
                for (int nt = 0; nt < 4; ++nt) o[u][nt][j] *= corr;
            }

            // C-layout -> A-layout transpose through LDS (in-order per wave)
#pragma unroll
            for (int j = 0; j < 8; ++j) {
                int mr = j + half_id * 8;
                pshare[mr * 32 + ln]      = (_Float16)p0[j];
                pshare[mr * 32 + 16 + ln] = (_Float16)p1[j];
            }
            __syncthreads();   // 1-wave WG: S_NOP + compiler ordering fence
            pa[u] = ld_a16(pshare + (size_t)ln * 32, kb8);
            __syncthreads();
        }

        // ---- O += P * V; each V tile feeds both query sub-tiles ----
#pragma unroll
        for (int nt = 0; nt < 4; ++nt) {
            int h = nt * 16 + ln;
            const _Float16* vp = Vt + ((size_t)b * HEAD + h) * CTX + key0 + half_id * 16;
            v16h vb = ld_b16(vp);
            o[0][nt] = wmma_f16(pa[0], vb, o[0][nt]);
            o[1][nt] = wmma_f16(pa[1], vb, o[1][nt]);
        }
    }

    // ---- epilogue: one reciprocal per row, packed multiplies, f32 stores ----
    float* ob = out + ((size_t)b * CTX + q0) * HEAD;
#pragma unroll
    for (int u = 0; u < 2; ++u)
#pragma unroll
        for (int j = 0; j < 8; ++j) {
            const float inv = 1.0f / lrow[u][j];
            const int r = u * 16 + j + half_id * 8;
#pragma unroll
            for (int nt = 0; nt < 4; ++nt)
                ob[(size_t)r * HEAD + nt * 16 + ln] = o[u][nt][j] * inv;
        }
}

// ---------------------------------------------------------------------------
extern "C" void kernel_launch(void* const* d_in, const int* in_sizes, int n_in,
                              void* d_out, int out_size, void* d_ws, size_t ws_size,
                              hipStream_t stream) {
    const float* x  = (const float*)d_in[0];
    const float* Wq = (const float*)d_in[1];
    const float* Wk = (const float*)d_in[2];
    const float* Wv = (const float*)d_in[3];
    float* out = (float*)d_out;

    char* ws = (char*)d_ws;
    _Float16* Wt = (_Float16*)ws;                                   // 3*64*1024 f16
    _Float16* Qh = (_Float16*)(ws + 3 * 65536 * sizeof(_Float16));  // [16][2048][64]
    _Float16* Kh = Qh + (size_t)BATCH * CTX * HEAD;
    _Float16* Vt = Kh + (size_t)BATCH * CTX * HEAD;                 // [16][64][2048]

    prep_weights<<<768, 256, 0, stream>>>(Wq, Wk, Wv, Wt);
    proj_kernel<<<3 * BATCH * (CTX / 64), 32, 0, stream>>>(x, Wt, Qh, Kh, Vt);
    attn_kernel<<<BATCH * (CTX / 32), 32, 0, stream>>>(Qh, Kh, Vt, out);
}
